// GatedRecurrence_19533511262472
// MI455X (gfx1250) — matrix operand order repaired and checked by reading, not compiled, not run on hardware
//
#include <hip/hip_runtime.h>
#include <hip/hip_bf16.h>
#include <cstdint>
#include <cstddef>

// ---------------------------------------------------------------------------
// Types
// ---------------------------------------------------------------------------
typedef __bf16 bf16_t;
typedef __attribute__((ext_vector_type(16))) __bf16        v16bf;
typedef __attribute__((ext_vector_type(8)))  float         v8f;
typedef __attribute__((ext_vector_type(4)))  unsigned int  v4u;
typedef __attribute__((ext_vector_type(8)))  int           v8i;
typedef __attribute__((ext_vector_type(4)))  int           v4i;

// Problem constants (from reference: B=4, T=4096, D=1024)
#define BB   4
#define TT   4096
#define DD   1024
#define BT   16384          // B*T rows
#define NG   3072           // 3*D gate columns
#define CHK  64             // scan chunk length
#define NCH  64             // chunks per sequence (T/CHK)

// ---------------------------------------------------------------------------
// Workspace layout (bytes)
// ---------------------------------------------------------------------------
#define OFF_QG   ((size_t)0)                           // 3072*1024 bf16 = 6 MiB
#define OFF_QO   (OFF_QG  + (size_t)NG*DD*2)           // 1024*1024 bf16 = 2 MiB
#define OFF_XHI  (OFF_QO  + (size_t)DD*DD*2)           // 16384*1024 bf16 = 32 MiB
#define OFF_XLO  (OFF_XHI + (size_t)BT*DD*2)
#define OFF_F    (OFF_XLO + (size_t)BT*DD*2)           // f      fp32 64 MiB
#define OFF_C    (OFF_F   + (size_t)BT*DD*4)           // silu(c) fp32 64 MiB
#define OFF_O    (OFF_C   + (size_t)BT*DD*4)           // o      fp32 64 MiB
#define OFF_AC   (OFF_O   + (size_t)BT*DD*4)           // chunk A / carry, 1 MiB
#define OFF_BC   (OFF_AC  + (size_t)BB*NCH*DD*4)       // chunk B, 1 MiB
#define OFF_ACC  (OFF_BC  + (size_t)BB*NCH*DD*4)       // 2 floats abs-sum
#define OFF_SCL  (OFF_ACC + (size_t)256)               // 2 floats scales

// ---------------------------------------------------------------------------
// TDM: issue a 2-D bf16 tile load (tile = 32 cols x 128 rows) into LDS.
// D# packing per CDNA5 ISA ch.8 (group0: count/lds/global/type, group1: dims).
// ---------------------------------------------------------------------------
__device__ __forceinline__ void tdm_load_tile_bf16(
    const void* gbase, unsigned long long elem_off, unsigned lds_byte,
    unsigned tdim0, unsigned tdim1, unsigned long long stride0_elems)
{
  unsigned long long ga = (unsigned long long)gbase + (elem_off << 1);
  v4u g0;
  g0.x = 1u;                                            // count=1, user D#
  g0.y = lds_byte;                                      // lds_addr (bytes)
  g0.z = (unsigned)ga;                                  // global_addr[31:0]
  g0.w = (unsigned)((ga >> 32) & 0x1FFFFFFull) | (2u << 30); // addr[56:32] | type=2
  v8i g1;
  g1[0] = (int)(1u << 16);                              // wg_mask=0, data_size=1 (2B)
  g1[1] = (int)((tdim0 & 0xFFFFu) << 16);               // tensor_dim0[15:0]
  g1[2] = (int)((tdim0 >> 16) | ((tdim1 & 0xFFFFu) << 16));
  g1[3] = (int)((tdim1 >> 16) | (32u << 16));           // tile_dim0 = 32 elems
  g1[4] = (int)128;                                     // tile_dim1 = 128 rows, tile_dim2=0
  g1[5] = (int)(unsigned)(stride0_elems & 0xFFFFFFFFull);
  g1[6] = (int)(unsigned)((stride0_elems >> 32) & 0xFFFFull);
  g1[7] = 0;
  v4i z4 = {0, 0, 0, 0};
#if defined(__clang_major__) && (__clang_major__ >= 23)
  v8i z8 = {0, 0, 0, 0, 0, 0, 0, 0};
  __builtin_amdgcn_tensor_load_to_lds(g0, g1, z4, z4, z8, 0);
#else
  __builtin_amdgcn_tensor_load_to_lds(g0, g1, z4, z4, 0);
#endif
}

// ---------------------------------------------------------------------------
// GEMM: C[M,N] = scale * (Ahi + Alo) @ Bw^T, A,Bw bf16 row-major [.,K].
// EPI==0: fused gate epilogue (f/silu(c)/o to three fp32 buffers, D=1024 cols)
// EPI==1: plain fp32 store to outY.
// Block 256 thr (8 waves): 128x128 tile, BK=32, double-buffered TDM into LDS.
// Wave (wm in 0..1, wn in 0..3) owns 64x32 -> acc[4][2] of 16x16 f32 tiles.
// ---------------------------------------------------------------------------
template <int EPI>
__global__ __launch_bounds__(256) void gemm_bf16_wmma(
    const bf16_t* __restrict__ Ahi, const bf16_t* __restrict__ Alo,
    const bf16_t* __restrict__ Bw,
    const float* __restrict__ scaleP, const float* __restrict__ bias,
    float* __restrict__ outF, float* __restrict__ outC, float* __restrict__ outO,
    float* __restrict__ outY, int M, int N, int K)
{
  extern __shared__ __align__(16) unsigned char smem[];
  const int tid  = threadIdx.x;
  const int lane = tid & 31;
  const int wave = tid >> 5;
  const int wm = wave >> 2;          // 0..1 -> 64-row slab
  const int wn = wave & 3;           // 0..3 -> 32-col slab
  const int mBlock = blockIdx.y * 128;
  const int nBlock = blockIdx.x * 128;
  const unsigned TILEB = 128 * 32 * 2;                 // 8 KiB per staged tile
  const unsigned ldsBase = (unsigned)(uintptr_t)smem;  // LDS byte offset (low 32 bits)

  v8f zero = {0.f, 0.f, 0.f, 0.f, 0.f, 0.f, 0.f, 0.f};
  v8f acc[4][2];
#pragma unroll
  for (int a = 0; a < 4; ++a) { acc[a][0] = zero; acc[a][1] = zero; }

  const int KT = K >> 5;
  if (tid < 32) {  // wave 0 drives the Tensor Data Mover
    tdm_load_tile_bf16(Ahi, (unsigned long long)mBlock * K, ldsBase + 0,         K, M, K);
    tdm_load_tile_bf16(Alo, (unsigned long long)mBlock * K, ldsBase + TILEB,     K, M, K);
    tdm_load_tile_bf16(Bw,  (unsigned long long)nBlock * K, ldsBase + 2 * TILEB, K, N, K);
  }

  for (int kt = 0; kt < KT; ++kt) {
    const int buf = kt & 1;
    if (tid < 32) {
      if (kt + 1 < KT) {
        const unsigned nb = (unsigned)(buf ^ 1) * 3u * TILEB;
        const unsigned long long ko = (unsigned long long)(kt + 1) * 32ull;
        tdm_load_tile_bf16(Ahi, (unsigned long long)mBlock * K + ko, ldsBase + nb + 0,         K, M, K);
        tdm_load_tile_bf16(Alo, (unsigned long long)mBlock * K + ko, ldsBase + nb + TILEB,     K, M, K);
        tdm_load_tile_bf16(Bw,  (unsigned long long)nBlock * K + ko, ldsBase + nb + 2 * TILEB, K, N, K);
        __builtin_amdgcn_s_wait_tensorcnt(3);  // current buffer's 3 loads done
      } else {
        __builtin_amdgcn_s_wait_tensorcnt(0);
      }
    }
    __syncthreads();

    const bf16_t* sAhi = (const bf16_t*)(smem + (size_t)buf * 3 * TILEB);
    const bf16_t* sAlo = sAhi + 128 * 32;
    const bf16_t* sB   = sAlo + 128 * 32;
    const int kHalfA = (lane >> 4) << 3;   // A: lanes>=16 own K base 8 (and +16)
    const int kHalfB = (lane >> 4) << 3;   // B: lanes>=16 own K base 8 (and +16)

    v16bf bfr[2];
#pragma unroll
    for (int tn = 0; tn < 2; ++tn) {       // B frags: row = N col, two 8-elem K chunks per lane
      const bf16_t* p = sB + ((wn * 32 + tn * 16 + (lane & 15)) << 5) + kHalfB;
      union { v16bf v; uint4 q[2]; } u;
      u.q[0] = *(const uint4*)(p);
      u.q[1] = *(const uint4*)(p + 16);
      bfr[tn] = u.v;
    }
#pragma unroll
    for (int tm = 0; tm < 4; ++tm) {       // A frags: two 8-elem K chunks per lane
      const int r = wm * 64 + tm * 16 + (lane & 15);
      union { v16bf v; uint4 q[2]; } uh, ul;
      const bf16_t* ph = sAhi + (r << 5) + kHalfA;
      uh.q[0] = *(const uint4*)(ph);
      uh.q[1] = *(const uint4*)(ph + 16);
      const bf16_t* pl = sAlo + (r << 5) + kHalfA;
      ul.q[0] = *(const uint4*)(pl);
      ul.q[1] = *(const uint4*)(pl + 16);
#pragma unroll
      for (int tn = 0; tn < 2; ++tn) {
        acc[tm][tn] = __builtin_amdgcn_wmma_f32_16x16x32_bf16(
            false, uh.v, false, bfr[tn], (short)0, acc[tm][tn], false, false);
        acc[tm][tn] = __builtin_amdgcn_wmma_f32_16x16x32_bf16(
            false, ul.v, false, bfr[tn], (short)0, acc[tm][tn], false, false);
      }
    }
    __syncthreads();   // all waves done with buf before TDM overwrites it
  }

  // Epilogue. C/D layout: lane L, vgpr i -> (M = i + 8*(L>=16), N = L%16).
  const float sc = scaleP[0];
#pragma unroll
  for (int tm = 0; tm < 4; ++tm) {
#pragma unroll
    for (int tn = 0; tn < 2; ++tn) {
      const int nCol  = nBlock + wn * 32 + tn * 16 + (lane & 15);
      const int mBase = mBlock + wm * 64 + tm * 16 + ((lane >> 4) << 3);
#pragma unroll
      for (int i = 0; i < 8; ++i) {
        const int mRow = mBase + i;
        const float g = acc[tm][tn][i] * sc;
        if (EPI == 0) {
          const int gate = nCol >> 10;        // tile never straddles a gate
          const int col  = nCol & 1023;
          const size_t idx = (size_t)mRow * DD + col;
          if (gate == 0) {
            outF[idx] = 1.f / (1.f + __expf(-(g + bias[col])));   // f
          } else if (gate == 1) {
            outC[idx] = g / (1.f + __expf(-g));                   // silu(c)
          } else {
            outO[idx] = 1.f / (1.f + __expf(-g));                 // o
          }
        } else {
          outY[(size_t)mRow * N + nCol] = g;
        }
      }
    }
  }
}

// ---------------------------------------------------------------------------
// Ternarization support
// ---------------------------------------------------------------------------
__global__ void absmean_partial(const float* __restrict__ w, long long n,
                                float* __restrict__ acc)
{
  __shared__ float red[256];
  float s = 0.f;
  for (long long i = (long long)blockIdx.x * blockDim.x + threadIdx.x; i < n;
       i += (long long)gridDim.x * blockDim.x)
    s += fabsf(w[i]);
  red[threadIdx.x] = s;
  __syncthreads();
  for (int k = 128; k > 0; k >>= 1) {
    if (threadIdx.x < k) red[threadIdx.x] += red[threadIdx.x + k];
    __syncthreads();
  }
  if (threadIdx.x == 0) atomicAdd(acc, red[0]);
}

__global__ void ternarize_pack(const float* __restrict__ w, bf16_t* __restrict__ q,
                               long long n, const float* __restrict__ acc,
                               float invN, float* __restrict__ scaleOut)
{
  const float scale = acc[0] * invN + 1e-8f;
  const long long i = (long long)blockIdx.x * blockDim.x + threadIdx.x;
  if (i < n) {
    float r = rintf(w[i] / scale);               // round-half-even like jnp.round
    r = fminf(1.f, fmaxf(-1.f, r));
    q[i] = (bf16_t)r;                            // exact in bf16: -1, 0, +1
  }
  if (i == 0) scaleOut[0] = scale;
}

__global__ void split_hilo(const float* __restrict__ x, bf16_t* __restrict__ hi,
                           bf16_t* __restrict__ lo, long long n)
{
  for (long long i = (long long)blockIdx.x * blockDim.x + threadIdx.x; i < n;
       i += (long long)gridDim.x * blockDim.x) {
    const float v = x[i];
    const bf16_t h = (bf16_t)v;
    hi[i] = h;
    lo[i] = (bf16_t)(v - (float)h);
  }
}

// ---------------------------------------------------------------------------
// Chunked linear scan: h[t] = f[t]*h[t-1] + (1-f[t])*c[t]   (fp32, per b,d)
// ---------------------------------------------------------------------------
__global__ void scan_chunks(const float* __restrict__ f, const float* __restrict__ c,
                            float* __restrict__ Ac, float* __restrict__ Bc)
{
  const long long id = (long long)blockIdx.x * blockDim.x + threadIdx.x; // B*NCH*D
  const int d  = (int)(id & (DD - 1));
  const int ch = (int)((id >> 10) & (NCH - 1));
  const int b  = (int)(id >> 16);
  const size_t base = ((size_t)b * TT + (size_t)ch * CHK) * DD + d;
  float A = 1.f, Bv = 0.f;
#pragma unroll 4
  for (int t = 0; t < CHK; ++t) {
    const float ft = f[base + (size_t)t * DD];
    const float ct = c[base + (size_t)t * DD];
    Bv = ft * Bv + (1.f - ft) * ct;
    A *= ft;
  }
  const size_t idx = ((size_t)b * NCH + ch) * DD + d;
  Ac[idx] = A;
  Bc[idx] = Bv;
}

__global__ void scan_carry(float* __restrict__ Ac, const float* __restrict__ Bc)
{
  const int id = blockIdx.x * blockDim.x + threadIdx.x;  // B*D = 4096
  const int d = id & (DD - 1);
  const int b = id >> 10;
  float h = 0.f;   // h[-1] = 0
  for (int ch = 0; ch < NCH; ++ch) {
    const size_t idx = ((size_t)b * NCH + ch) * DD + d;
    const float a = Ac[idx], bb = Bc[idx];
    Ac[idx] = h;                  // exclusive carry replaces A in-place
    h = a * h + bb;
  }
}

__global__ void scan_apply(const float* __restrict__ f, const float* __restrict__ c,
                           const float* __restrict__ o, const float* __restrict__ carry,
                           bf16_t* __restrict__ hi, bf16_t* __restrict__ lo)
{
  const long long id = (long long)blockIdx.x * blockDim.x + threadIdx.x;
  const int d  = (int)(id & (DD - 1));
  const int ch = (int)((id >> 10) & (NCH - 1));
  const int b  = (int)(id >> 16);
  const size_t base = ((size_t)b * TT + (size_t)ch * CHK) * DD + d;
  float h = carry[((size_t)b * NCH + ch) * DD + d];
#pragma unroll 4
  for (int t = 0; t < CHK; ++t) {
    const size_t p = base + (size_t)t * DD;
    const float ft = f[p];
    h = ft * h + (1.f - ft) * c[p];
    const float y = h * o[p];            // fuse output gate
    const bf16_t yh = (bf16_t)y;
    hi[p] = yh;
    lo[p] = (bf16_t)(y - (float)yh);     // hi/lo split for GEMM2
  }
}

// ---------------------------------------------------------------------------
// Launch
// ---------------------------------------------------------------------------
extern "C" void kernel_launch(void* const* d_in, const int* in_sizes, int n_in,
                              void* d_out, int out_size, void* d_ws, size_t ws_size,
                              hipStream_t stream)
{
  (void)in_sizes; (void)n_in; (void)out_size; (void)ws_size;
  const float* x     = (const float*)d_in[0];   // (4,4096,1024) fp32
  const float* wg    = (const float*)d_in[1];   // (3072,1024)   fp32
  const float* wo    = (const float*)d_in[2];   // (1024,1024)   fp32
  const float* fbias = (const float*)d_in[3];   // (1024,)       fp32

  char* ws = (char*)d_ws;
  bf16_t* qg   = (bf16_t*)(ws + OFF_QG);
  bf16_t* qo   = (bf16_t*)(ws + OFF_QO);
  bf16_t* xhi  = (bf16_t*)(ws + OFF_XHI);   // reused as (h*o)_hi after GEMM1
  bf16_t* xlo  = (bf16_t*)(ws + OFF_XLO);   // reused as (h*o)_lo
  float*  fbuf = (float*)(ws + OFF_F);
  float*  cbuf = (float*)(ws + OFF_C);
  float*  obuf = (float*)(ws + OFF_O);
  float*  Ac   = (float*)(ws + OFF_AC);
  float*  Bc   = (float*)(ws + OFF_BC);
  float*  accG = (float*)(ws + OFF_ACC);
  float*  accO = accG + 1;
  float*  sclG = (float*)(ws + OFF_SCL);
  float*  sclO = sclG + 1;

  hipMemsetAsync(accG, 0, 2 * sizeof(float), stream);

  absmean_partial<<<1024, 256, 0, stream>>>(wg, (long long)NG * DD, accG);
  absmean_partial<<<512,  256, 0, stream>>>(wo, (long long)DD * DD, accO);

  ternarize_pack<<<(NG * DD + 255) / 256, 256, 0, stream>>>(
      wg, qg, (long long)NG * DD, accG, 1.f / ((float)NG * DD), sclG);
  ternarize_pack<<<(DD * DD + 255) / 256, 256, 0, stream>>>(
      wo, qo, (long long)DD * DD, accO, 1.f / ((float)DD * DD), sclO);

  split_hilo<<<4096, 256, 0, stream>>>(x, xhi, xlo, (long long)BT * DD);

  const size_t lds_bytes = 2u * 3u * 128u * 32u * 2u;  // 48 KiB double-buffered
  dim3 grid1(NG / 128, BT / 128);                      // (24, 128)
  gemm_bf16_wmma<0><<<grid1, 256, lds_bytes, stream>>>(
      xhi, xlo, qg, sclG, fbias, fbuf, cbuf, obuf, nullptr, BT, NG, DD);

  scan_chunks<<<(BB * NCH * DD) / 256, 256, 0, stream>>>(fbuf, cbuf, Ac, Bc);
  scan_carry<<<(BB * DD) / 256, 256, 0, stream>>>(Ac, Bc);
  scan_apply<<<(BB * NCH * DD) / 256, 256, 0, stream>>>(fbuf, cbuf, obuf, Ac, xhi, xlo);

  dim3 grid2(DD / 128, BT / 128);                      // (8, 128)
  gemm_bf16_wmma<1><<<grid2, 256, lds_bytes, stream>>>(
      xhi, xlo, qo, sclO, nullptr, nullptr, nullptr, nullptr,
      (float*)d_out, BT, DD, DD);
}
